// LibragradMultiScaleAttention_24060406792852
// MI455X (gfx1250) — matrix operand order repaired and checked by reading, not compiled, not run on hardware
//
#include <hip/hip_runtime.h>
#include <hip/hip_bf16.h>

typedef __attribute__((ext_vector_type(16))) _Float16 v16h;
typedef __attribute__((ext_vector_type(8)))  _Float16 v8h;
typedef __attribute__((ext_vector_type(4)))  _Float16 v4h;
typedef __attribute__((ext_vector_type(8)))  float    v8f;

#define N_TOK 4096
#define DIM   768
#define QKV_D 2304
#define NH    12
#define HD    64

// ---------------------------------------------------------------------------
// D = A(16x32 f16) * B(32x16 f16) + C(16x16 f32), wave32 WMMA
// ---------------------------------------------------------------------------
__device__ __forceinline__ v8f wmma_f16(v16h a, v16h b, v8f c) {
  return __builtin_amdgcn_wmma_f32_16x16x32_f16(
      /*neg_a=*/false, a, /*neg_b=*/false, b,
      /*c_mod=*/(short)0, c, /*reuse_a=*/false, /*reuse_b=*/false);
}

// A-operand fragment: 16 rows x 32 K, row-major source with stride in halves.
// Contiguous k order (same as B): lanes 0-15 -> M=lane, K=0..15; lanes 16-31 -> M=lane-16,
// K=16..31. Two 16B contiguous loads per lane.
__device__ __forceinline__ v16h load_a_frag(const _Float16* __restrict__ p, int stride_h) {
  const int lane = threadIdx.x & 31;
  const int row  = lane & 15;
  const int kb   = (lane >> 4) << 4;   // 0 or 16
  const _Float16* r = p + row * stride_h;
  v8h lo = *(const v8h*)(r + kb);        // K = kb .. kb+7
  v8h hi = *(const v8h*)(r + kb + 8);    // K = kb+8 .. kb+15
  v16h out;
#pragma unroll
  for (int i = 0; i < 8; ++i) { out[i] = lo[i]; out[8 + i] = hi[i]; }
  return out;
}

// B-operand fragment: 32 K x 16 N, source stored N-major (i.e. B^T row-major),
// stride in halves. Contiguous k order (same as A): lanes 0-15 -> N=lane, K=0..15; lanes 16-31 ->
// N=lane-16, K=16..31. One 32B contiguous load per lane.
__device__ __forceinline__ v16h load_b_frag(const _Float16* __restrict__ p, int stride_h) {
  const int lane = threadIdx.x & 31;
  const int col  = lane & 15;
  const int kb   = (lane >> 4) << 4;   // 0 or 16
  return *(const v16h*)(p + col * stride_h + kb);
}

// ---------------------------------------------------------------------------
// fp32 -> fp16 conversion (vectorized x4)
// ---------------------------------------------------------------------------
__global__ void __launch_bounds__(256)
cvt_f32_to_f16(const float* __restrict__ in, _Float16* __restrict__ out, int n4) {
  int i = blockIdx.x * 256 + threadIdx.x;
  if (i < n4) {
    float4 v = ((const float4*)in)[i];
    v4h h;
    h[0] = (_Float16)v.x; h[1] = (_Float16)v.y;
    h[2] = (_Float16)v.z; h[3] = (_Float16)v.w;
    ((v4h*)out)[i] = h;
  }
}

// ---------------------------------------------------------------------------
// GEMM1: qkv[n,d] = x[n,:] . qkv_w[d,:] + qkv_b[d]
// 128x128 tile per block, 8 waves, each wave 64(M)x32(N).
// Epilogue scatters into head-major q, k and TRANSPOSED v (vT[h][e][n]).
// ---------------------------------------------------------------------------
__global__ void __launch_bounds__(256)
gemm_qkv(const _Float16* __restrict__ xh, const _Float16* __restrict__ wh,
         const float* __restrict__ bias,
         _Float16* __restrict__ qh, _Float16* __restrict__ kh,
         _Float16* __restrict__ vth) {
  const int wave = threadIdx.x >> 5;
  const int wm = wave & 1, wn = wave >> 1;
  const int m0 = blockIdx.x * 128 + wm * 64;
  const int n0 = blockIdx.y * 128 + wn * 32;

  v8f acc[4][2] = {};
  for (int k0 = 0; k0 < DIM; k0 += 32) {
    v16h a[4], b[2];
#pragma unroll
    for (int i = 0; i < 4; ++i) a[i] = load_a_frag(xh + (m0 + i * 16) * DIM + k0, DIM);
#pragma unroll
    for (int j = 0; j < 2; ++j) b[j] = load_b_frag(wh + (n0 + j * 16) * DIM + k0, DIM);
#pragma unroll
    for (int i = 0; i < 4; ++i)
#pragma unroll
      for (int j = 0; j < 2; ++j) acc[i][j] = wmma_f16(a[i], b[j], acc[i][j]);
  }

  const int lane = threadIdx.x & 31;
  const int nc = lane & 15;
  const int mo = (lane >> 4) << 3;
  const int mat = n0 / DIM;            // 0=q, 1=k, 2=v (uniform per block)
  const int nrem0 = n0 - mat * DIM;
#pragma unroll
  for (int j = 0; j < 2; ++j) {
    const int d   = n0 + j * 16 + nc;
    const float bv = bias[d];
    const int rem = nrem0 + j * 16 + nc;
    const int hh  = rem >> 6;
    const int e   = rem & 63;
#pragma unroll
    for (int i = 0; i < 4; ++i)
#pragma unroll
      for (int r = 0; r < 8; ++r) {
        const int n = m0 + i * 16 + mo + r;
        const _Float16 hv = (_Float16)(acc[i][j][r] + bv);
        if (mat == 0)      qh [(hh * N_TOK + n) * HD + e] = hv;
        else if (mat == 1) kh [(hh * N_TOK + n) * HD + e] = hv;
        else               vth[(hh * HD + e) * N_TOK + n] = hv;
      }
  }
}

// ---------------------------------------------------------------------------
// Flash attention: block = one head x 64 q-rows; 4 waves x 16 q-rows each.
// S = q.k^T (WMMA, fp32 acc), online softmax in fp32, O += P.V (WMMA).
// P is re-laid out C-layout -> A-layout through per-wave private LDS.
// ---------------------------------------------------------------------------
__global__ void __launch_bounds__(128)
attn_fa(const _Float16* __restrict__ qh, const _Float16* __restrict__ kh,
        const _Float16* __restrict__ vth, _Float16* __restrict__ aoh) {
  __shared__ _Float16 plds[4][16 * 72];   // 72-half row pitch: 16B aligned, bank-spread

  const int h    = blockIdx.y;
  const int wave = threadIdx.x >> 5;
  const int lane = threadIdx.x & 31;
  const int qrow0 = blockIdx.x * 64 + wave * 16;

  // q A-fragments for this wave's 16 rows (K = 64 -> two fragments), kept in VGPRs
  const _Float16* qbase = qh + (h * N_TOK + qrow0) * HD;
  v16h qa0 = load_a_frag(qbase + 0,  HD);
  v16h qa1 = load_a_frag(qbase + 32, HD);

  float mrow[8], lrow[8];
#pragma unroll
  for (int r = 0; r < 8; ++r) { mrow[r] = -1e30f; lrow[r] = 0.0f; }
  v8f o[4] = {};
  const float scale = 0.125f;  // 64^-0.5
  _Float16* pw = &plds[wave][0];

  for (int kt = 0; kt < N_TOK / 64; ++kt) {
    const _Float16* kbase = kh  + (h * N_TOK + kt * 64) * HD;
    const _Float16* vbase = vth + (size_t)h * HD * N_TOK + kt * 64;
    if (kt + 1 < N_TOK / 64) {
      __builtin_prefetch(kbase + 64 * HD, 0, 1);
      __builtin_prefetch(vbase + 64, 0, 1);
    }

    // S(16x64) = q(16x64) . k^T   (4 column tiles x 2 K-steps)
    v8f s[4] = {};
#pragma unroll
    for (int j = 0; j < 4; ++j) {
      v16h b0 = load_b_frag(kbase + (j * 16) * HD + 0,  HD);
      v16h b1 = load_b_frag(kbase + (j * 16) * HD + 32, HD);
      s[j] = wmma_f16(qa0, b0, s[j]);
      s[j] = wmma_f16(qa1, b1, s[j]);
    }
#pragma unroll
    for (int j = 0; j < 4; ++j)
#pragma unroll
      for (int r = 0; r < 8; ++r) s[j][r] *= scale;

    // online softmax (row = C-layout VGPR index; cols spread over 16 lanes x 4 tiles)
#pragma unroll
    for (int r = 0; r < 8; ++r) {
      float mx = fmaxf(fmaxf(s[0][r], s[1][r]), fmaxf(s[2][r], s[3][r]));
#pragma unroll
      for (int off = 8; off >= 1; off >>= 1) mx = fmaxf(mx, __shfl_xor(mx, off, 16));
      const float mnew = fmaxf(mrow[r], mx);
      float srow = 0.0f;
#pragma unroll
      for (int j = 0; j < 4; ++j) {
        const float p = __expf(s[j][r] - mnew);
        s[j][r] = p;
        srow += p;
      }
#pragma unroll
      for (int off = 8; off >= 1; off >>= 1) srow += __shfl_xor(srow, off, 16);
      const float alpha = __expf(mrow[r] - mnew);
      lrow[r] = lrow[r] * alpha + srow;
      mrow[r] = mnew;
#pragma unroll
      for (int j = 0; j < 4; ++j) o[j][r] *= alpha;
    }

    // P: C-layout -> A-layout via per-wave LDS scratch (no workgroup barrier;
    // same-wave cross-lane RAW covered by s_wait_dscnt)
    {
      const int nc = lane & 15;
      const int mo = (lane >> 4) << 3;
#pragma unroll
      for (int j = 0; j < 4; ++j)
#pragma unroll
        for (int r = 0; r < 8; ++r)
          pw[(mo + r) * 72 + j * 16 + nc] = (_Float16)s[j][r];
    }
    asm volatile("s_wait_dscnt 0x0" ::: "memory");

    // O(16x64) += P(16x64) . V(64x64)  — B fragments stream from vT (N-major)
    v16h pa0 = load_a_frag(pw + 0,  72);
    v16h pa1 = load_a_frag(pw + 32, 72);
#pragma unroll
    for (int j = 0; j < 4; ++j) {
      v16h b0 = load_b_frag(vbase + (j * 16) * N_TOK + 0,  N_TOK);
      v16h b1 = load_b_frag(vbase + (j * 16) * N_TOK + 32, N_TOK);
      o[j] = wmma_f16(pa0, b0, o[j]);
      o[j] = wmma_f16(pa1, b1, o[j]);
    }
  }

  // epilogue: normalize and store fp16 into (n, h*64+e) layout for proj GEMM
  const int nc = lane & 15;
  const int mo = (lane >> 4) << 3;
#pragma unroll
  for (int j = 0; j < 4; ++j)
#pragma unroll
    for (int r = 0; r < 8; ++r) {
      const int n = qrow0 + mo + r;
      const int e = j * 16 + nc;
      aoh[n * DIM + h * HD + e] = (_Float16)(o[j][r] / lrow[r]);
    }
}

// ---------------------------------------------------------------------------
// GEMM2: out[n,d] = ao[n,:] . proj_w[d,:] + proj_b[d]   (fp32 output)
// ---------------------------------------------------------------------------
__global__ void __launch_bounds__(256)
gemm_proj(const _Float16* __restrict__ aoh, const _Float16* __restrict__ pwh,
          const float* __restrict__ bias, float* __restrict__ out) {
  const int wave = threadIdx.x >> 5;
  const int wm = wave & 1, wn = wave >> 1;
  const int m0 = blockIdx.x * 128 + wm * 64;
  const int n0 = blockIdx.y * 128 + wn * 32;

  v8f acc[4][2] = {};
  for (int k0 = 0; k0 < DIM; k0 += 32) {
    v16h a[4], b[2];
#pragma unroll
    for (int i = 0; i < 4; ++i) a[i] = load_a_frag(aoh + (m0 + i * 16) * DIM + k0, DIM);
#pragma unroll
    for (int j = 0; j < 2; ++j) b[j] = load_b_frag(pwh + (n0 + j * 16) * DIM + k0, DIM);
#pragma unroll
    for (int i = 0; i < 4; ++i)
#pragma unroll
      for (int j = 0; j < 2; ++j) acc[i][j] = wmma_f16(a[i], b[j], acc[i][j]);
  }

  const int lane = threadIdx.x & 31;
  const int nc = lane & 15;
  const int mo = (lane >> 4) << 3;
#pragma unroll
  for (int j = 0; j < 2; ++j) {
    const int d = n0 + j * 16 + nc;
    const float bv = bias[d];
#pragma unroll
    for (int i = 0; i < 4; ++i)
#pragma unroll
      for (int r = 0; r < 8; ++r)
        out[(m0 + i * 16 + mo + r) * DIM + d] = acc[i][j][r] + bv;
  }
}

// ---------------------------------------------------------------------------
extern "C" void kernel_launch(void* const* d_in, const int* in_sizes, int n_in,
                              void* d_out, int out_size, void* d_ws, size_t ws_size,
                              hipStream_t stream) {
  const float* x      = (const float*)d_in[0];  // (1,64,64,768)
  const float* qkv_w  = (const float*)d_in[1];  // (2304,768)
  const float* qkv_b  = (const float*)d_in[2];  // (2304,)
  const float* proj_w = (const float*)d_in[3];  // (768,768)
  const float* proj_b = (const float*)d_in[4];  // (768,)
  float* out = (float*)d_out;

  char* ws = (char*)d_ws;
  _Float16* xh     = (_Float16*)(ws + 0);          // 4096*768*2   = 6,291,456
  _Float16* qkvwh  = (_Float16*)(ws + 6291456);    // 2304*768*2   = 3,538,944
  _Float16* projwh = (_Float16*)(ws + 9830400);    // 768*768*2    = 1,179,648
  _Float16* qh     = (_Float16*)(ws + 11010048);   // 12*4096*64*2 = 6,291,456
  _Float16* kh     = (_Float16*)(ws + 17301504);   // 6,291,456
  _Float16* vth    = (_Float16*)(ws + 23592960);   // 6,291,456 (transposed per head)
  _Float16* aoh    = (_Float16*)(ws + 29884416);   // 6,291,456  -> total ~36.2 MB

  cvt_f32_to_f16<<<dim3((N_TOK * DIM / 4 + 255) / 256), 256, 0, stream>>>(x, xh, N_TOK * DIM / 4);
  cvt_f32_to_f16<<<dim3((QKV_D * DIM / 4 + 255) / 256), 256, 0, stream>>>(qkv_w, qkvwh, QKV_D * DIM / 4);
  cvt_f32_to_f16<<<dim3((DIM * DIM / 4 + 255) / 256), 256, 0, stream>>>(proj_w, projwh, DIM * DIM / 4);

  gemm_qkv<<<dim3(N_TOK / 128, QKV_D / 128), 256, 0, stream>>>(xh, qkvwh, qkv_b, qh, kh, vth);
  attn_fa<<<dim3(N_TOK / 64, NH), 128, 0, stream>>>(qh, kh, vth, aoh);
  gemm_proj<<<dim3(N_TOK / 128, DIM / 128), 256, 0, stream>>>(aoh, projwh, proj_b, out);
}
